// cross_attention_layer_48893907698163
// MI455X (gfx1250) — matrix operand order repaired and checked by reading, not compiled, not run on hardware
//
#include <hip/hip_runtime.h>
#include <hip/hip_bf16.h>
#include <math.h>

// Problem constants (match reference)
#define BB 32
#define NN 1024
#define MM 1024
#define DD 128

typedef __bf16 bf16;
typedef __attribute__((ext_vector_type(16))) __bf16 v16bf;
typedef __attribute__((ext_vector_type(2)))  __bf16 bf16x2;
typedef __attribute__((ext_vector_type(8)))  float  v8f;

#define DEV static __device__ __forceinline__

// ---- WMMA fragment helpers (wave32, v_wmma_f32_16x16x32_bf16 layouts) ----
// A (16x32, 16-bit): lane L holds row M = L%16; VGPR v holds K pair:
//   K = (v>=4 ? 16 : 0) + (L/16)*8 + (v&3)*2
// B (32x16, 16-bit): lane L holds col N = L%16; VGPR v holds K pair:
//   K = (v>=4 ? 16 : 0) + (L/16)*8 + (v&3)*2
// C/D (16x16 f32): lane L col N = L%16; VGPR r row M = 8*(L/16) + r

DEV v16bf load_a_bf16_row(const bf16* __restrict__ row, int k0, int half) {
  v16bf a;
#pragma unroll
  for (int v = 0; v < 8; ++v) {
    int kp = k0 + ((v & 4) ? 16 : 0) + half * 8 + (v & 3) * 2;
    bf16x2 p = *(const bf16x2*)(row + kp);
    a[2 * v] = p.x; a[2 * v + 1] = p.y;
  }
  return a;
}

// B(k,n) where element (k,n) sits at colbase[k] (contiguous in K for this
// lane's column). Used both for "transpose-of-row-major" operands (Y^T from
// row-major Y) and for pre-transposed staging buffers (YT / XT).
DEV v16bf load_b_bf16_contig(const bf16* __restrict__ colbase, int k0, int half) {
  v16bf b;
#pragma unroll
  for (int v = 0; v < 8; ++v) {
    int k = k0 + ((v & 4) ? 16 : 0) + half * 8 + (v & 3) * 2;
    bf16x2 p = *(const bf16x2*)(colbase + k);
    b[2 * v] = p.x; b[2 * v + 1] = p.y;
  }
  return b;
}

DEV v8f wmma_bf16(v16bf a, v16bf b, v8f c) {
  return __builtin_amdgcn_wmma_f32_16x16x32_bf16(false, a, false, b, (short)0, c,
                                                 false, false);
}

// ---------------------------------------------------------------------------
// K1: coeff = tanh(W @ A_w^T + A_b); Xc = bf16(X * coeff) row-major;
//     XT = bf16(X) transposed per batch: XT[b][d][n].
// GEMM over flat rows [B*N, D] x A_w^T[D, D]; one 16x16 tile per wave.
// ---------------------------------------------------------------------------
__global__ void coeff_xc_kernel(const float* __restrict__ W,
                                const float* __restrict__ Xs,
                                const float* __restrict__ Aw,
                                const float* __restrict__ Ab,
                                bf16* __restrict__ Xc, bf16* __restrict__ XT) {
  const int wave = (blockIdx.x * blockDim.x + threadIdx.x) >> 5;
  const int lane = threadIdx.x & 31;
  const int half = lane >> 4, l16 = lane & 15;
  const int te = wave & (DD / 16 - 1);        // 8 e-tiles
  const int tr = wave >> 3;                   // row tile over B*N
  const int arow = tr * 16 + l16;             // A-fragment row (lane-private)
  const int ecol = te * 16 + l16;             // B-fragment col (lane-private)

  const float* wrow = W + (size_t)arow * DD;
  const float* acol = Aw + (size_t)ecol * DD; // B(k,n) = Aw[n*D + k]
  v8f acc = {};
#pragma unroll
  for (int kt = 0; kt < DD / 32; ++kt) {
    const int k0 = kt * 32;
    v16bf a, bm;
#pragma unroll
    for (int v = 0; v < 8; ++v) {
      int kp = k0 + ((v & 4) ? 16 : 0) + half * 8 + (v & 3) * 2;
      float2 p = *(const float2*)(wrow + kp);
      a[2 * v] = (bf16)p.x; a[2 * v + 1] = (bf16)p.y;
      int kb = k0 + ((v & 4) ? 16 : 0) + half * 8 + (v & 3) * 2;
      float2 q = *(const float2*)(acol + kb);
      bm[2 * v] = (bf16)q.x; bm[2 * v + 1] = (bf16)q.y;
    }
    acc = wmma_bf16(a, bm, acc);
  }
  const float bias = Ab[ecol];
#pragma unroll
  for (int r = 0; r < 8; ++r) {
    const int orow = tr * 16 + half * 8 + r;        // flat row = b*N + n
    const int ob = orow >> 10, on = orow & (NN - 1);
    const size_t idx = (size_t)orow * DD + ecol;
    const float c  = tanhf(acc[r] + bias);
    const float xv = Xs[idx];
    Xc[idx] = (bf16)(xv * c);
    XT[((size_t)ob * DD + ecol) * NN + on] = (bf16)xv;
  }
}

// K2: Yh = bf16(Ys) row-major; YT = bf16(Ys) transposed [b][d][m].
__global__ void convert_y_kernel(const float* __restrict__ Ys,
                                 bf16* __restrict__ Yh, bf16* __restrict__ YT) {
  const int i = blockIdx.x * blockDim.x + threadIdx.x;   // pair index
  float2 p = ((const float2*)Ys)[i];
  bf16x2 q; q.x = (bf16)p.x; q.y = (bf16)p.y;
  ((bf16x2*)Yh)[i] = q;
  const int elem = 2 * i;
  const int b = elem / (MM * DD);
  const int rem = elem - b * (MM * DD);
  const int m = rem / DD, d = rem - m * DD;
  YT[((size_t)b * DD + d) * MM + m]     = q.x;
  YT[((size_t)b * DD + d + 1) * MM + m] = q.y;
}

// ---------------------------------------------------------------------------
// K3: s = softplus(Xc @ Y^T) - 0.5  -> f32 [B,N,M]
// 16x64 output tile per wave: one A fragment feeds 4 WMMAs per K-step.
// ---------------------------------------------------------------------------
__global__ void affinity_kernel(const bf16* __restrict__ Xc,
                                const bf16* __restrict__ Yh,
                                float* __restrict__ s) {
  const int wave = (blockIdx.x * blockDim.x + threadIdx.x) >> 5;
  const int lane = threadIdx.x & 31;
  const int half = lane >> 4, l16 = lane & 15;
  const int TPB = (NN / 16) * (MM / 64);      // 1024 tiles per batch
  const int b = wave / TPB, rem = wave % TPB;
  const int tn = rem / (MM / 64), tm4 = rem % (MM / 64);

  const bf16* arow = Xc + ((size_t)b * NN + tn * 16 + l16) * DD;
  const bf16* brow[4];
#pragma unroll
  for (int j = 0; j < 4; ++j)   // B(k,n) = Y[n*D + k] : contiguous in k
    brow[j] = Yh + ((size_t)b * MM + tm4 * 64 + j * 16 + l16) * DD;

  v8f acc[4] = {};
#pragma unroll
  for (int kt = 0; kt < DD / 32; ++kt) {
    v16bf a = load_a_bf16_row(arow, kt * 32, half);
#pragma unroll
    for (int j = 0; j < 4; ++j) {
      v16bf bm = load_b_bf16_contig(brow[j], kt * 32, half);
      acc[j] = wmma_bf16(a, bm, acc[j]);
    }
  }
#pragma unroll
  for (int j = 0; j < 4; ++j) {
#pragma unroll
    for (int r = 0; r < 8; ++r) {
      const int n = tn * 16 + half * 8 + r;
      const int m = tm4 * 64 + j * 16 + l16;
      const float x  = acc[j][r];
      const float sp = (x > 20.f) ? x : log1pf(__expf(x));
      s[((size_t)b * NN + n) * MM + m] = sp - 0.5f;
    }
  }
}

// K4: per-row (axis=2) max & sum(exp) — one wave32 per row of s
__global__ void row_stats_kernel(const float* __restrict__ s,
                                 float* __restrict__ rmax, float* __restrict__ rsum) {
  const int row = (blockIdx.x * blockDim.x + threadIdx.x) >> 5; // in [0, B*N)
  const int lane = threadIdx.x & 31;
  const float* p = s + (size_t)row * MM;
  float mx = -3.0e38f;
  for (int j = lane; j < MM; j += 32) mx = fmaxf(mx, p[j]);
#pragma unroll
  for (int o = 16; o > 0; o >>= 1) mx = fmaxf(mx, __shfl_xor(mx, o, 32));
  float sm = 0.f;
  for (int j = lane; j < MM; j += 32) sm += __expf(p[j] - mx);
#pragma unroll
  for (int o = 16; o > 0; o >>= 1) sm += __shfl_xor(sm, o, 32);
  if (lane == 0) { rmax[row] = mx; rsum[row] = sm; }
}

// K5: per-column (axis=1) max & sum(exp) — one wave32 per (b,m)
__global__ void col_stats_kernel(const float* __restrict__ s,
                                 float* __restrict__ cmax, float* __restrict__ csum) {
  const int col = (blockIdx.x * blockDim.x + threadIdx.x) >> 5; // in [0, B*M)
  const int lane = threadIdx.x & 31;
  const int b = col / MM, m = col % MM;
  const float* p = s + (size_t)b * NN * MM + m;
  float mx = -3.0e38f;
  for (int j = lane; j < NN; j += 32) mx = fmaxf(mx, p[(size_t)j * MM]);
#pragma unroll
  for (int o = 16; o > 0; o >>= 1) mx = fmaxf(mx, __shfl_xor(mx, o, 32));
  float sm = 0.f;
  for (int j = lane; j < NN; j += 32) sm += __expf(p[(size_t)j * MM] - mx);
#pragma unroll
  for (int o = 16; o > 0; o >>= 1) sm += __shfl_xor(sm, o, 32);
  if (lane == 0) { cmax[col] = mx; csum[col] = sm; }
}

// ---------------------------------------------------------------------------
// K6: attention_x = softmax_row(s) @ Y.  A = exp(s - rowmax) built on the fly
// (bf16); one A fragment feeds 4 WMMAs (16x64 of D); divide by rowsum at end.
// B operand from transposed staging YT[b][d][m] -> contiguous-in-K loads.
// ---------------------------------------------------------------------------
__global__ void attn_x_kernel(const float* __restrict__ s,
                              const float* __restrict__ rmax,
                              const float* __restrict__ rsum,
                              const bf16* __restrict__ YT,
                              float* __restrict__ outx) {
  const int wave = (blockIdx.x * blockDim.x + threadIdx.x) >> 5;
  const int lane = threadIdx.x & 31;
  const int half = lane >> 4, l16 = lane & 15;
  const int TPB = (NN / 16) * (DD / 64);      // 128 tiles per batch
  const int b = wave / TPB, rem = wave % TPB;
  const int tn = rem / (DD / 64), dt = rem % (DD / 64);

  const int n = tn * 16 + l16;                // A-fragment row
  const float* srow = s + ((size_t)b * NN + n) * MM;
  const float rm = rmax[(size_t)b * NN + n];
  const bf16* ytcol[4];
#pragma unroll
  for (int j = 0; j < 4; ++j) {               // B(k,d) = YT[b][d][k]
    const int d = dt * 64 + j * 16 + l16;
    ytcol[j] = YT + ((size_t)b * DD + d) * MM;
  }

  v8f acc[4] = {};
  for (int kt = 0; kt < MM / 32; ++kt) {
    const int k0 = kt * 32;
    __builtin_prefetch(srow + k0 + 32, 0, 0);
    v16bf a;
#pragma unroll
    for (int v = 0; v < 8; ++v) {
      int kp = k0 + ((v & 4) ? 16 : 0) + half * 8 + (v & 3) * 2;
      float2 p = *(const float2*)(srow + kp);
      a[2 * v]     = (bf16)__expf(p.x - rm);
      a[2 * v + 1] = (bf16)__expf(p.y - rm);
    }
#pragma unroll
    for (int j = 0; j < 4; ++j) {
      v16bf bm = load_b_bf16_contig(ytcol[j], k0, half);
      acc[j] = wmma_bf16(a, bm, acc[j]);
    }
  }
#pragma unroll
  for (int r = 0; r < 8; ++r) {
    const int orow = tn * 16 + half * 8 + r;
    const float inv = 1.0f / rsum[(size_t)b * NN + orow];
#pragma unroll
    for (int j = 0; j < 4; ++j) {
      const int d = dt * 64 + j * 16 + l16;
      outx[((size_t)b * NN + orow) * DD + d] = acc[j][r] * inv;
    }
  }
}

// ---------------------------------------------------------------------------
// K7: attention_y = softmax_col(s)^T @ X.  A rows are m, K runs over n
// (transposed access into s, stride M); B from XT[b][d][n] (contiguous in K).
// ---------------------------------------------------------------------------
__global__ void attn_y_kernel(const float* __restrict__ s,
                              const float* __restrict__ cmax,
                              const float* __restrict__ csum,
                              const bf16* __restrict__ XT,
                              float* __restrict__ outy) {
  const int wave = (blockIdx.x * blockDim.x + threadIdx.x) >> 5;
  const int lane = threadIdx.x & 31;
  const int half = lane >> 4, l16 = lane & 15;
  const int TPB = (MM / 16) * (DD / 64);      // 128 tiles per batch
  const int b = wave / TPB, rem = wave % TPB;
  const int tm = rem / (DD / 64), dt = rem % (DD / 64);

  const int m = tm * 16 + l16;                // A-fragment row
  const float* scol = s + (size_t)b * NN * MM + m; // element (k=n, m) at scol[k*M]
  const float cm = cmax[(size_t)b * MM + m];
  const bf16* xtcol[4];
#pragma unroll
  for (int j = 0; j < 4; ++j) {               // B(k,d) = XT[b][d][k]
    const int d = dt * 64 + j * 16 + l16;
    xtcol[j] = XT + ((size_t)b * DD + d) * NN;
  }

  v8f acc[4] = {};
  for (int kt = 0; kt < NN / 32; ++kt) {
    const int k0 = kt * 32;
    __builtin_prefetch(scol + (size_t)(k0 + 32) * MM, 0, 0);
    v16bf a;
#pragma unroll
    for (int v = 0; v < 8; ++v) {
      int kp = k0 + ((v & 4) ? 16 : 0) + half * 8 + (v & 3) * 2;
      a[2 * v]     = (bf16)__expf(scol[(size_t)kp * MM] - cm);
      a[2 * v + 1] = (bf16)__expf(scol[(size_t)(kp + 1) * MM] - cm);
    }
#pragma unroll
    for (int j = 0; j < 4; ++j) {
      v16bf bm = load_b_bf16_contig(xtcol[j], k0, half);
      acc[j] = wmma_bf16(a, bm, acc[j]);
    }
  }
#pragma unroll
  for (int r = 0; r < 8; ++r) {
    const int orow = tm * 16 + half * 8 + r;
    const float inv = 1.0f / csum[(size_t)b * MM + orow];
#pragma unroll
    for (int j = 0; j < 4; ++j) {
      const int d = dt * 64 + j * 16 + l16;
      outy[((size_t)b * MM + orow) * DD + d] = acc[j][r] * inv;
    }
  }
}

// ---------------------------------------------------------------------------
extern "C" void kernel_launch(void* const* d_in, const int* in_sizes, int n_in,
                              void* d_out, int out_size, void* d_ws, size_t ws_size,
                              hipStream_t stream) {
  (void)in_sizes; (void)n_in; (void)out_size; (void)ws_size;
  const float* Xs = (const float*)d_in[0];
  const float* Ys = (const float*)d_in[1];
  const float* W  = (const float*)d_in[2];
  const float* Aw = (const float*)d_in[3];
  const float* Ab = (const float*)d_in[4];

  float* outx = (float*)d_out;                         // [B,N,D]
  float* outy = outx + (size_t)BB * NN * DD;           // [B,M,D]
  float* s    = outy + (size_t)BB * MM * DD;           // [B,N,M]

  // Workspace layout (needs ~32.5 MB)
  char* ws = (char*)d_ws;
  bf16* Xc = (bf16*)(ws);                                   // 8 MB  [B,N,D]
  bf16* XT = (bf16*)(ws + (size_t)BB * NN * DD * 2);        // 8 MB  [B,D,N]
  bf16* Yh = (bf16*)(ws + (size_t)BB * NN * DD * 4);        // 8 MB  [B,M,D]
  bf16* YT = (bf16*)(ws + (size_t)BB * NN * DD * 6);        // 8 MB  [B,D,M]
  float* rmax = (float*)(ws + (size_t)BB * NN * DD * 8);
  float* rsum = rmax + (size_t)BB * NN;
  float* cmax = rsum + (size_t)BB * NN;
  float* csum = cmax + (size_t)BB * MM;

  const int THREADS = 128;                // 4 wave32 per block
  // K1: (B*N/16) * (D/16) waves
  {
    int waves = (BB * NN / 16) * (DD / 16);
    coeff_xc_kernel<<<waves / 4, THREADS, 0, stream>>>(W, Xs, Aw, Ab, Xc, XT);
  }
  // K2: Y -> bf16 (+ transpose), 2 elems/thread
  {
    int n2 = BB * MM * DD / 2;
    convert_y_kernel<<<n2 / 256, 256, 0, stream>>>(Ys, Yh, YT);
  }
  // K3: affinity, 16x64 tiles
  {
    int waves = BB * (NN / 16) * (MM / 64);
    affinity_kernel<<<waves / 4, THREADS, 0, stream>>>(Xc, Yh, s);
  }
  // K4/K5: softmax stats
  {
    int rows = BB * NN;
    row_stats_kernel<<<rows / 4, THREADS, 0, stream>>>(s, rmax, rsum);
    int cols = BB * MM;
    col_stats_kernel<<<cols / 4, THREADS, 0, stream>>>(s, cmax, csum);
  }
  // K6/K7: attention GEMMs, 16x64 tiles
  {
    int waves = BB * (NN / 16) * (DD / 64);
    attn_x_kernel<<<waves / 4, THREADS, 0, stream>>>(s, rmax, rsum, YT, outx);
    int wavesy = BB * (MM / 16) * (DD / 64);
    attn_y_kernel<<<wavesy / 4, THREADS, 0, stream>>>(s, cmax, csum, XT, outy);
  }
}
